// HyperGraphSynergy_30167850287712
// MI455X (gfx1250) — compile-verified, matrix operand order checked by reading, not run on hardware
//
#include <hip/hip_runtime.h>
#include <hip/hip_bf16.h>

typedef __bf16 bf16;
typedef __attribute__((ext_vector_type(8)))  bf16  v8bf;
typedef __attribute__((ext_vector_type(16))) bf16  v16bf;
typedef __attribute__((ext_vector_type(8)))  float v8f;

#define N_ATOMS   1520
#define NUM_DRUGS 38
#define N_CLINE   4096
#define N_GRAPH   4134
#define N_HYPER   20000
#define E_DRUG    6080
#define E_HYP     60000
#define N_IDX     200000

#define ACT_NONE  0
#define ACT_RELU  1
#define ACT_LEAKY 2
#define ACT_TANH  3
#define ACT_SIGM  4

__device__ __forceinline__ float act_apply(float v, int act) {
  switch (act) {
    case ACT_RELU:  return v > 0.f ? v : 0.f;
    case ACT_LEAKY: return v > 0.f ? v : 0.2f * v;
    case ACT_TANH:  return tanhf(v);
    case ACT_SIGM:  return 1.f / (1.f + expf(-v));
  }
  return v;
}

// ---------------------------------------------------------------------------
// WMMA 16x16x32 bf16 tile machinery (wave32). Block = 256 thr = 8 waves.
// Block tile 128(M) x 64(N); wave grid 4(M) x 2(N); wave tile 32x32
// (2 A-frags x 2 B-frags -> 4 WMMAs, frag reuse). Double-buffered LDS,
// global loads for tile k+1 issued before computing tile k.
// ---------------------------------------------------------------------------
__device__ __forceinline__ void mma_step(const bf16 (*As)[32], const bf16 (*Bs)[32],
                                         int wm, int wn, int lane, v8f acc[2][2]) {
  const int half = lane >> 4, l16 = lane & 15;
  v16bf a[2], b[2];
#pragma unroll
  for (int ma = 0; ma < 2; ++ma) {
    const int am = wm * 32 + ma * 16 + l16;
    v8bf alo = *(const v8bf*)&As[am][half * 8];
    v8bf ahi = *(const v8bf*)&As[am][16 + half * 8];
#pragma unroll
    for (int i = 0; i < 8; ++i) { a[ma][i] = alo[i]; a[ma][i + 8] = ahi[i]; }
  }
#pragma unroll
  for (int nb = 0; nb < 2; ++nb) {
    const int bn = wn * 32 + nb * 16 + l16;
    v8bf blo = *(const v8bf*)&Bs[bn][half * 8];
    v8bf bhi = *(const v8bf*)&Bs[bn][16 + half * 8];
#pragma unroll
    for (int i = 0; i < 8; ++i) { b[nb][i] = blo[i]; b[nb][i + 8] = bhi[i]; }
  }
#pragma unroll
  for (int ma = 0; ma < 2; ++ma)
#pragma unroll
    for (int nb = 0; nb < 2; ++nb)
      acc[ma][nb] = __builtin_amdgcn_wmma_f32_16x16x32_bf16(
          false, a[ma], false, b[nb], (short)0, acc[ma][nb], false, false);
}

__device__ __forceinline__ void epilogue(v8f acc[2][2], int wm, int wn, int lane,
                                         int rowBase, int colBase, int M, int N,
                                         const float* bias, int act,
                                         float* outF, int ldc, bf16* outB, int ldob) {
  const int half = lane >> 4, l16 = lane & 15;
#pragma unroll
  for (int ma = 0; ma < 2; ++ma) {
#pragma unroll
    for (int nb = 0; nb < 2; ++nb) {
      const int col = colBase + wn * 32 + nb * 16 + l16;
      if (col >= N) continue;
      const float badd = bias ? bias[col] : 0.f;
#pragma unroll
      for (int r = 0; r < 8; ++r) {
        const int row = rowBase + wm * 32 + ma * 16 + r + 8 * half;
        if (row >= M) continue;
        float v = acc[ma][nb][r] + badd;
        v = act_apply(v, act);
        if (outF) outF[(size_t)row * ldc + col] = v;
        if (outB) outB[(size_t)row * ldob + col] = (bf16)v;
      }
    }
  }
}

// C[M,N] = A[M,K] * BT[N,K]^T  (both bf16 row-major, K % 32 == 0)
__global__ void k_wmma_nt(const bf16* __restrict__ A, int lda,
                          const bf16* __restrict__ BT, int ldb,
                          const float* __restrict__ aScale, const float* __restrict__ aShift,
                          const float* __restrict__ bias,
                          float* __restrict__ outF, int ldc,
                          bf16* __restrict__ outB, int ldob,
                          int M, int N, int K, int act) {
  __shared__ __align__(16) bf16 As[2][128][32];
  __shared__ __align__(16) bf16 Bs[2][64][32];
  const int t = threadIdx.x;
  const int lane = t & 31, wave = t >> 5;
  const int wm = wave >> 1, wn = wave & 1;
  const int rowBase = blockIdx.y * 128;
  const int colBase = blockIdx.x * 64;
  const int aRow = t >> 1,  aSeg = (t & 1) << 4;   // 128 rows x 32 cols (2 x v8bf)
  const int bRow = t >> 2,  bSeg = (t & 3) << 3;   // 64 rows x 32 cols (1 x v8bf)

  v8f acc[2][2];
#pragma unroll
  for (int ma = 0; ma < 2; ++ma)
#pragma unroll
    for (int nb = 0; nb < 2; ++nb)
#pragma unroll
      for (int i = 0; i < 8; ++i) acc[ma][nb][i] = 0.f;

  v8bf ra0, ra1, rb;
  auto loadRegs = [&](int k0) {
#pragma unroll
    for (int i = 0; i < 8; ++i) { ra0[i] = (bf16)0.f; ra1[i] = (bf16)0.f; rb[i] = (bf16)0.f; }
    const int gr = rowBase + aRow;
    if (gr < M) {
      const bf16* ap = A + (size_t)gr * lda + k0 + aSeg;
      ra0 = *(const v8bf*)ap;
      ra1 = *(const v8bf*)(ap + 8);
      if (aScale) {
#pragma unroll
        for (int i = 0; i < 8; ++i) {
          const int kc = k0 + aSeg + i;
          ra0[i] = (bf16)((float)ra0[i] * aScale[kc] + aShift[kc]);
          ra1[i] = (bf16)((float)ra1[i] * aScale[kc + 8] + aShift[kc + 8]);
        }
      }
    }
    const int gn = colBase + bRow;
    if (gn < N) rb = *(const v8bf*)(BT + (size_t)gn * ldb + k0 + bSeg);
  };
  auto storeLDS = [&](int p) {
    *(v8bf*)&As[p][aRow][aSeg]     = ra0;
    *(v8bf*)&As[p][aRow][aSeg + 8] = ra1;
    *(v8bf*)&Bs[p][bRow][bSeg]     = rb;
  };

  loadRegs(0);
  storeLDS(0);
  int p = 0;
  for (int k0 = 0; k0 < K; k0 += 32) {
    const bool more = (k0 + 32) < K;
    if (more) loadRegs(k0 + 32);        // global loads in flight during compute
    __syncthreads();                    // LDS buffer p is ready
    mma_step(As[p], Bs[p], wm, wn, lane, acc);
    if (more) storeLDS(1 - p);
    p ^= 1;
  }
  epilogue(acc, wm, wn, lane, rowBase, colBase, M, N, bias, act, outF, ldc, outB, ldob);
}

// Gather variant: row i of A = [G[idx[i,0]] | G[idx[i,1]] | G[idx[i,2]]], K = 3*ldg
__global__ void k_wmma_gather3(const bf16* __restrict__ G, int ldg,
                               const int* __restrict__ idx,
                               const bf16* __restrict__ BT, int ldb,
                               const float* __restrict__ bias,
                               bf16* __restrict__ outB, int ldob,
                               int M, int N, int act) {
  __shared__ __align__(16) bf16 As[2][128][32];
  __shared__ __align__(16) bf16 Bs[2][64][32];
  const int t = threadIdx.x;
  const int lane = t & 31, wave = t >> 5;
  const int wm = wave >> 1, wn = wave & 1;
  const int rowBase = blockIdx.y * 128;
  const int colBase = blockIdx.x * 64;
  const int aRow = t >> 1,  aSeg = (t & 1) << 4;
  const int bRow = t >> 2,  bSeg = (t & 3) << 3;
  const int K = 3 * ldg;

  v8f acc[2][2];
#pragma unroll
  for (int ma = 0; ma < 2; ++ma)
#pragma unroll
    for (int nb = 0; nb < 2; ++nb)
#pragma unroll
      for (int i = 0; i < 8; ++i) acc[ma][nb][i] = 0.f;

  v8bf ra0, ra1, rb;
  auto loadRegs = [&](int k0) {
#pragma unroll
    for (int i = 0; i < 8; ++i) { ra0[i] = (bf16)0.f; ra1[i] = (bf16)0.f; rb[i] = (bf16)0.f; }
    const int gr = rowBase + aRow;
    if (gr < M) {
      const int part = k0 / ldg;
      const int kin = (k0 - part * ldg) + aSeg;
      const int src = idx[(size_t)gr * 3 + part];
      const bf16* ap = G + (size_t)src * ldg + kin;
      ra0 = *(const v8bf*)ap;
      ra1 = *(const v8bf*)(ap + 8);
    }
    const int gn = colBase + bRow;
    if (gn < N) rb = *(const v8bf*)(BT + (size_t)gn * ldb + k0 + bSeg);
  };
  auto storeLDS = [&](int p) {
    *(v8bf*)&As[p][aRow][aSeg]     = ra0;
    *(v8bf*)&As[p][aRow][aSeg + 8] = ra1;
    *(v8bf*)&Bs[p][bRow][bSeg]     = rb;
  };

  loadRegs(0);
  storeLDS(0);
  int p = 0;
  for (int k0 = 0; k0 < K; k0 += 32) {
    const bool more = (k0 + 32) < K;
    if (more) loadRegs(k0 + 32);
    __syncthreads();
    mma_step(As[p], Bs[p], wm, wn, lane, acc);
    if (more) storeLDS(1 - p);
    p ^= 1;
  }
  epilogue(acc, wm, wn, lane, rowBase, colBase, M, N, bias, act,
           (float*)nullptr, 0, outB, ldob);
}

// ---------------------------------------------------------------------------
// Elementwise / reduction / scatter helpers
// ---------------------------------------------------------------------------
__global__ void k_fill_f32(float* p, float v, long n) {
  for (long i = blockIdx.x * (long)blockDim.x + threadIdx.x; i < n;
       i += (long)gridDim.x * blockDim.x) p[i] = v;
}
__global__ void k_fill_u32(unsigned* p, unsigned v, long n) {
  for (long i = blockIdx.x * (long)blockDim.x + threadIdx.x; i < n;
       i += (long)gridDim.x * blockDim.x) p[i] = v;
}
__global__ void k_cvt_pad(const float* __restrict__ src, bf16* __restrict__ dst,
                          int M, int K, int Kp) {
  const long n = (long)M * Kp;
  for (long i = blockIdx.x * (long)blockDim.x + threadIdx.x; i < n;
       i += (long)gridDim.x * blockDim.x) {
    const int r = (int)(i / Kp), k = (int)(i % Kp);
    dst[i] = (bf16)((k < K) ? src[(size_t)r * K + k] : 0.f);
  }
}
__global__ void k_transpose_pad(const float* __restrict__ W, bf16* __restrict__ WT,
                                int K, int N, int Kp) {
  const long n = (long)N * Kp;
  for (long i = blockIdx.x * (long)blockDim.x + threadIdx.x; i < n;
       i += (long)gridDim.x * blockDim.x) {
    const int c = (int)(i / Kp), k = (int)(i % Kp);
    WT[i] = (bf16)((k < K) ? W[(size_t)k * N + c] : 0.f);
  }
}
__global__ void k_count(const int* __restrict__ idx, float* __restrict__ cnt, int E) {
  for (int i = blockIdx.x * blockDim.x + threadIdx.x; i < E;
       i += gridDim.x * blockDim.x) atomicAdd(&cnt[idx[i]], 1.f);
}
__global__ void k_rsqrt_p1(float* d, int n) {
  for (int i = blockIdx.x * blockDim.x + threadIdx.x; i < n;
       i += gridDim.x * blockDim.x) d[i] = rsqrtf(d[i] + 1.f);  // +1 self loop
}
__global__ void k_recip(float* d, int n) {
  for (int i = blockIdx.x * blockDim.x + threadIdx.x; i < n;
       i += gridDim.x * blockDim.x) { float v = d[i]; d[i] = v > 0.f ? 1.f / v : 0.f; }
}
__global__ void k_gcn_self(const float* __restrict__ H, const float* __restrict__ dinv,
                           const float* __restrict__ b, float* __restrict__ out,
                           int M, int F) {
  const long n = (long)M * F;
  for (long i = blockIdx.x * (long)blockDim.x + threadIdx.x; i < n;
       i += (long)gridDim.x * blockDim.x) {
    const int r = (int)(i / F), c = (int)(i % F);
    const float di = dinv[r];
    out[i] = b[c] + H[i] * di * di;
  }
}
__global__ void k_gcn_edge(const int* __restrict__ src, const int* __restrict__ dst,
                           const float* __restrict__ H, const float* __restrict__ dinv,
                           float* __restrict__ out, int E, int F) {
  const long n = (long)E * F;
  for (long i = blockIdx.x * (long)blockDim.x + threadIdx.x; i < n;
       i += (long)gridDim.x * blockDim.x) {
    const int e = (int)(i / F), c = (int)(i % F);
    const int s = src[e], d = dst[e];
    atomicAdd(&out[(size_t)d * F + c], H[(size_t)s * F + c] * dinv[s] * dinv[d]);
  }
}
__global__ void k_act(float* x, long n, int act) {
  for (long i = blockIdx.x * (long)blockDim.x + threadIdx.x; i < n;
       i += (long)gridDim.x * blockDim.x) x[i] = act_apply(x[i], act);
}
__global__ void k_colstats_f32(const float* __restrict__ X, int ld,
                               float* __restrict__ sum, float* __restrict__ sq, int M) {
  const int f = blockIdx.x;
  float s = 0.f, ss = 0.f;
  for (int r = blockIdx.y * blockDim.x + threadIdx.x; r < M;
       r += gridDim.y * blockDim.x) {
    const float v = X[(size_t)r * ld + f]; s += v; ss += v * v;
  }
  __shared__ float shs[256], shq[256];
  const int t = threadIdx.x;
  shs[t] = s; shq[t] = ss; __syncthreads();
  for (int o = 128; o > 0; o >>= 1) {
    if (t < o) { shs[t] += shs[t + o]; shq[t] += shq[t + o]; }
    __syncthreads();
  }
  if (t == 0) { atomicAdd(&sum[f], shs[0]); atomicAdd(&sq[f], shq[0]); }
}
__global__ void k_colstats_bf16(const bf16* __restrict__ X, int ld,
                                float* __restrict__ sum, float* __restrict__ sq, int M) {
  const int f = blockIdx.x;
  float s = 0.f, ss = 0.f;
  for (int r = blockIdx.y * blockDim.x + threadIdx.x; r < M;
       r += gridDim.y * blockDim.x) {
    const float v = (float)X[(size_t)r * ld + f]; s += v; ss += v * v;
  }
  __shared__ float shs[256], shq[256];
  const int t = threadIdx.x;
  shs[t] = s; shq[t] = ss; __syncthreads();
  for (int o = 128; o > 0; o >>= 1) {
    if (t < o) { shs[t] += shs[t + o]; shq[t] += shq[t + o]; }
    __syncthreads();
  }
  if (t == 0) { atomicAdd(&sum[f], shs[0]); atomicAdd(&sq[f], shq[0]); }
}
__global__ void k_finstats(const float* sum, const float* sq, const float* g,
                           const float* b, float* scale, float* shift,
                           int F, float invM) {
  const int f = blockIdx.x * blockDim.x + threadIdx.x;
  if (f >= F) return;
  const float m = sum[f] * invM;
  const float v = sq[f] * invM - m * m;
  const float sc = g[f] * rsqrtf(v + 1e-5f);
  scale[f] = sc;
  shift[f] = b[f] - m * sc;
}
__global__ void k_bn(const float* __restrict__ X, int ld, const float* scale,
                     const float* shift, float* outF, bf16* outB, int M, int F) {
  const long n = (long)M * F;
  for (long i = blockIdx.x * (long)blockDim.x + threadIdx.x; i < n;
       i += (long)gridDim.x * blockDim.x) {
    const int r = (int)(i / F), c = (int)(i % F);
    const float v = X[(size_t)r * ld + c] * scale[c] + shift[c];
    if (outF) outF[(size_t)r * F + c] = v;
    if (outB) outB[(size_t)r * F + c] = (bf16)v;
  }
}
__device__ __forceinline__ unsigned fenc(float f) {
  const unsigned u = __float_as_uint(f);
  return (u & 0x80000000u) ? ~u : (u | 0x80000000u);
}
__device__ __forceinline__ float fdec(unsigned u) {
  u = (u & 0x80000000u) ? (u & 0x7FFFFFFFu) : ~u;
  return __uint_as_float(u);
}
__global__ void k_segmax_enc(const float* __restrict__ X, const int* __restrict__ seg,
                             unsigned* __restrict__ keys, int M, int F) {
  const long n = (long)M * F;
  for (long i = blockIdx.x * (long)blockDim.x + threadIdx.x; i < n;
       i += (long)gridDim.x * blockDim.x) {
    const int r = (int)(i / F), c = (int)(i % F);
    atomicMax(&keys[(size_t)seg[r] * F + c], fenc(X[i]));
  }
}
__global__ void k_segmax_dec(const unsigned* __restrict__ keys, float* __restrict__ out,
                             long n) {
  for (long i = blockIdx.x * (long)blockDim.x + threadIdx.x; i < n;
       i += (long)gridDim.x * blockDim.x) {
    float v = fdec(keys[i]);
    if (!(fabsf(v) < 3.0e38f)) v = 0.f;   // non-finite -> 0
    out[i] = v;
  }
}
__global__ void k_merge(const float* __restrict__ demb, const float* __restrict__ cel,
                        bf16* __restrict__ out, int F, int Kp) {
  const long n = (long)N_GRAPH * Kp;
  for (long i = blockIdx.x * (long)blockDim.x + threadIdx.x; i < n;
       i += (long)gridDim.x * blockDim.x) {
    const int r = (int)(i / Kp), k = (int)(i % Kp);
    float v = 0.f;
    if (k < F) v = (r < NUM_DRUGS) ? demb[(size_t)r * F + k]
                                   : cel[(size_t)(r - NUM_DRUGS) * F + k];
    out[i] = (bf16)v;
  }
}
// dst[b[e]][:] += src[a[e]][:]
__global__ void k_hscat(const int* __restrict__ a, const int* __restrict__ b,
                        const float* __restrict__ src, float* __restrict__ dst,
                        int E, int F) {
  const long n = (long)E * F;
  for (long i = blockIdx.x * (long)blockDim.x + threadIdx.x; i < n;
       i += (long)gridDim.x * blockDim.x) {
    const int e = (int)(i / F), c = (int)(i % F);
    atomicAdd(&dst[(size_t)b[e] * F + c], src[(size_t)a[e] * F + c]);
  }
}
__global__ void k_scalerows(float* X, const float* s, int M, int F) {
  const long n = (long)M * F;
  for (long i = blockIdx.x * (long)blockDim.x + threadIdx.x; i < n;
       i += (long)gridDim.x * blockDim.x) X[i] *= s[i / F];
}
__global__ void k_hfinish(float* X, const float* __restrict__ dinv,
                          const float* __restrict__ b, int M, int F) {
  const long n = (long)M * F;
  for (long i = blockIdx.x * (long)blockDim.x + threadIdx.x; i < n;
       i += (long)gridDim.x * blockDim.x) {
    const int r = (int)(i / F), c = (int)(i % F);
    const float v = X[i] * dinv[r] + b[c];
    X[i] = v > 0.f ? v : 0.2f * v;
  }
}
// res[i] = sum_c (bn(z2[i,c])) * w[c] + b ; one wave per row (192 cols)
__global__ void k_final_dot(const bf16* __restrict__ z2, const float* __restrict__ scale,
                            const float* __restrict__ shift, const float* __restrict__ w,
                            const float* __restrict__ b, float* __restrict__ res, int M) {
  const int row = blockIdx.x * 8 + (threadIdx.x >> 5);
  const int lane = threadIdx.x & 31;
  if (row >= M) return;
  float s = 0.f;
  for (int c = lane; c < 192; c += 32) {
    const float v = (float)z2[(size_t)row * 192 + c] * scale[c] + shift[c];
    s += v * w[c];
  }
  for (int off = 16; off > 0; off >>= 1) s += __shfl_down(s, off, 32);
  if (lane == 0) res[row] = s + b[0];
}

// ---------------------------------------------------------------------------
extern "C" void kernel_launch(void* const* d_in, const int* in_sizes, int n_in,
                              void* d_out, int out_size, void* d_ws, size_t ws_size,
                              hipStream_t stream) {
  (void)in_sizes; (void)n_in; (void)out_size; (void)ws_size;
  const float* drug_x = (const float*)d_in[0];
  const float* gexpr  = (const float*)d_in[1];
  const int*   dei    = (const int*)d_in[2];
  const int*   ibatch = (const int*)d_in[3];
  const int*   hedge  = (const int*)d_in[4];
  const int*   indexp = (const int*)d_in[5];
  const float *Wg1=(const float*)d_in[6],  *bg1=(const float*)d_in[7];
  const float *g1g=(const float*)d_in[8],  *g1b=(const float*)d_in[9];
  const float *Wg2=(const float*)d_in[10], *bg2=(const float*)d_in[11];
  const float *g2g=(const float*)d_in[12], *g2b=(const float*)d_in[13];
  const float *Wc1=(const float*)d_in[14], *bc1=(const float*)d_in[15];
  const float *cg =(const float*)d_in[16], *cb =(const float*)d_in[17];
  const float *Wc2=(const float*)d_in[18], *bc2=(const float*)d_in[19];
  const float *Wh1=(const float*)d_in[20], *bh1=(const float*)d_in[21];
  const float *hg =(const float*)d_in[22], *hb =(const float*)d_in[23];
  const float *Wh2=(const float*)d_in[24], *bh2=(const float*)d_in[25];
  const float *Wrd=(const float*)d_in[26], *Wrc=(const float*)d_in[27];
  const float *Wd1=(const float*)d_in[28], *bd1=(const float*)d_in[29];
  const float *d1g=(const float*)d_in[30], *d1b=(const float*)d_in[31];
  const float *Wd2=(const float*)d_in[32], *bd2=(const float*)d_in[33];
  const float *d2g=(const float*)d_in[34], *d2b=(const float*)d_in[35];
  const float *Wd3=(const float*)d_in[36], *bd3=(const float*)d_in[37];
  float* out = (float*)d_out;
  float* out_res = out;                 // 200000
  float* out_rd  = out + 200000;        // 38*38
  float* out_rc  = out + 201444;        // 4096*4096

  size_t off = 0;
  auto alloc = [&](size_t bytes) -> void* {
    void* p = (char*)d_ws + off;
    off += (bytes + 255) & ~(size_t)255;
    return p;
  };
  auto cdiv = [](long a, long b) { return (int)((a + b - 1) / b); };
  auto gs = [](long n) { long b = (n + 255) / 256; if (b > 16384) b = 16384; if (b < 1) b = 1; return (int)b; };

  // workspace
  bf16* bf_drugx = (bf16*)alloc((size_t)N_ATOMS * 96 * 2);
  bf16* wtg1     = (bf16*)alloc(128 * 96 * 2);
  bf16* wtg2     = (bf16*)alloc(100 * 128 * 2);
  bf16* gexpr_bf = (bf16*)alloc((size_t)N_CLINE * 672 * 2);
  bf16* wtc1     = (bf16*)alloc(128 * 672 * 2);
  bf16* wtc2     = (bf16*)alloc(100 * 128 * 2);
  bf16* wth1     = (bf16*)alloc(256 * 128 * 2);
  bf16* wth2     = (bf16*)alloc(256 * 256 * 2);
  bf16* wtrd     = (bf16*)alloc(256 * 256 * 2);
  bf16* wtrc     = (bf16*)alloc(256 * 256 * 2);
  bf16* wtd1     = (bf16*)alloc((size_t)384 * 768 * 2);
  bf16* wtd2     = (bf16*)alloc((size_t)192 * 384 * 2);
  float* H1   = (float*)alloc((size_t)N_ATOMS * 128 * 4);
  float* agg1 = (float*)alloc((size_t)N_ATOMS * 128 * 4);
  bf16*  X1bf = (bf16*)alloc((size_t)N_ATOMS * 128 * 2);
  float* H2   = (float*)alloc((size_t)N_ATOMS * 100 * 4);
  float* agg2 = (float*)alloc((size_t)N_ATOMS * 100 * 4);
  float* deg  = (float*)alloc(N_ATOMS * 4);
  unsigned* keys = (unsigned*)alloc((size_t)NUM_DRUGS * 100 * 4);
  float* demb = (float*)alloc((size_t)NUM_DRUGS * 100 * 4);
  float* C1   = (float*)alloc((size_t)N_CLINE * 128 * 4);
  bf16*  C1bf = (bf16*)alloc((size_t)N_CLINE * 128 * 2);
  float* cellb= (float*)alloc((size_t)N_CLINE * 100 * 4);
  bf16*  mergebf = (bf16*)alloc((size_t)N_GRAPH * 128 * 2);
  float* Dd   = (float*)alloc(N_GRAPH * 4);
  float* Bd   = (float*)alloc(N_HYPER * 4);
  float* HH   = (float*)alloc((size_t)N_GRAPH * 256 * 4);
  float* msg  = (float*)alloc((size_t)N_HYPER * 256 * 4);
  float* aggH = (float*)alloc((size_t)N_GRAPH * 256 * 4);
  bf16*  hbf  = (bf16*)alloc((size_t)N_GRAPH * 256 * 2);
  float* aggG = (float*)alloc((size_t)N_GRAPH * 256 * 4);
  bf16*  gbf  = (bf16*)alloc((size_t)N_GRAPH * 256 * 2);
  float* Td   = (float*)alloc((size_t)NUM_DRUGS * 256 * 4);
  bf16*  Tdbf = (bf16*)alloc((size_t)NUM_DRUGS * 256 * 2);
  float* Tc   = (float*)alloc((size_t)N_CLINE * 256 * 4);
  bf16*  Tcbf = (bf16*)alloc((size_t)N_CLINE * 256 * 2);
  bf16*  z1   = (bf16*)alloc((size_t)N_IDX * 384 * 2);
  bf16*  z2   = (bf16*)alloc((size_t)N_IDX * 192 * 2);
  float* sumb = (float*)alloc(384 * 4);
  float* sqb  = (float*)alloc(384 * 4);
  float* scl  = (float*)alloc(384 * 4);
  float* shf  = (float*)alloc(384 * 4);

  auto gemm = [&](const bf16* A, int lda, const bf16* BT, int ldb,
                  const float* aS, const float* aH, const float* bias,
                  float* oF, int ldc, bf16* oB, int ldob,
                  int M, int N, int K, int act) {
    dim3 g(cdiv(N, 64), cdiv(M, 128));
    k_wmma_nt<<<g, 256, 0, stream>>>(A, lda, BT, ldb, aS, aH, bias,
                                     oF, ldc, oB, ldob, M, N, K, act);
  };
  auto bnstats_f32 = [&](const float* X, int ld, int M, int F,
                         const float* g, const float* b) {
    k_fill_f32<<<gs(F), 256, 0, stream>>>(sumb, 0.f, F);
    k_fill_f32<<<gs(F), 256, 0, stream>>>(sqb, 0.f, F);
    dim3 gr(F, 32);
    k_colstats_f32<<<gr, 256, 0, stream>>>(X, ld, sumb, sqb, M);
    k_finstats<<<cdiv(F, 256), 256, 0, stream>>>(sumb, sqb, g, b, scl, shf, F, 1.f / M);
  };
  auto bnstats_bf16 = [&](const bf16* X, int ld, int M, int F,
                          const float* g, const float* b) {
    k_fill_f32<<<gs(F), 256, 0, stream>>>(sumb, 0.f, F);
    k_fill_f32<<<gs(F), 256, 0, stream>>>(sqb, 0.f, F);
    dim3 gr(F, 64);
    k_colstats_bf16<<<gr, 256, 0, stream>>>(X, ld, sumb, sqb, M);
    k_finstats<<<cdiv(F, 256), 256, 0, stream>>>(sumb, sqb, g, b, scl, shf, F, 1.f / M);
  };

  // ---- conversions / weight transposes
  k_cvt_pad<<<gs((long)N_ATOMS * 96), 256, 0, stream>>>(drug_x, bf_drugx, N_ATOMS, 75, 96);
  k_transpose_pad<<<gs(128 * 96), 256, 0, stream>>>(Wg1, wtg1, 75, 128, 96);
  k_transpose_pad<<<gs(100 * 128), 256, 0, stream>>>(Wg2, wtg2, 128, 100, 128);
  k_cvt_pad<<<gs((long)N_CLINE * 672), 256, 0, stream>>>(gexpr, gexpr_bf, N_CLINE, 651, 672);
  k_transpose_pad<<<gs(128 * 672), 256, 0, stream>>>(Wc1, wtc1, 651, 128, 672);
  k_transpose_pad<<<gs(100 * 128), 256, 0, stream>>>(Wc2, wtc2, 128, 100, 128);
  k_transpose_pad<<<gs(256 * 128), 256, 0, stream>>>(Wh1, wth1, 100, 256, 128);
  k_transpose_pad<<<gs(256 * 256), 256, 0, stream>>>(Wh2, wth2, 256, 256, 256);
  k_transpose_pad<<<gs(256 * 256), 256, 0, stream>>>(Wrd, wtrd, 256, 256, 256);
  k_transpose_pad<<<gs(256 * 256), 256, 0, stream>>>(Wrc, wtrc, 256, 256, 256);
  k_transpose_pad<<<gs((long)384 * 768), 256, 0, stream>>>(Wd1, wtd1, 768, 384, 768);
  k_transpose_pad<<<gs((long)192 * 384), 256, 0, stream>>>(Wd2, wtd2, 384, 192, 384);

  // ---- drug graph degrees
  const int* e_src = dei;
  const int* e_dst = dei + E_DRUG;
  k_fill_f32<<<gs(N_ATOMS), 256, 0, stream>>>(deg, 0.f, N_ATOMS);
  k_count<<<gs(E_DRUG), 256, 0, stream>>>(e_dst, deg, E_DRUG);
  k_rsqrt_p1<<<gs(N_ATOMS), 256, 0, stream>>>(deg, N_ATOMS);   // deg -> dinv

  // ---- GCN layer 1
  gemm(bf_drugx, 96, wtg1, 96, nullptr, nullptr, nullptr, H1, 128, nullptr, 0,
       N_ATOMS, 128, 96, ACT_NONE);
  k_gcn_self<<<gs((long)N_ATOMS * 128), 256, 0, stream>>>(H1, deg, bg1, agg1, N_ATOMS, 128);
  k_gcn_edge<<<gs((long)E_DRUG * 128), 256, 0, stream>>>(e_src, e_dst, H1, deg, agg1, E_DRUG, 128);
  k_act<<<gs((long)N_ATOMS * 128), 256, 0, stream>>>(agg1, (long)N_ATOMS * 128, ACT_RELU);
  bnstats_f32(agg1, 128, N_ATOMS, 128, g1g, g1b);
  k_bn<<<gs((long)N_ATOMS * 128), 256, 0, stream>>>(agg1, 128, scl, shf, nullptr, X1bf, N_ATOMS, 128);

  // ---- GCN layer 2
  gemm(X1bf, 128, wtg2, 128, nullptr, nullptr, nullptr, H2, 100, nullptr, 0,
       N_ATOMS, 100, 128, ACT_NONE);
  k_gcn_self<<<gs((long)N_ATOMS * 100), 256, 0, stream>>>(H2, deg, bg2, agg2, N_ATOMS, 100);
  k_gcn_edge<<<gs((long)E_DRUG * 100), 256, 0, stream>>>(e_src, e_dst, H2, deg, agg2, E_DRUG, 100);
  k_act<<<gs((long)N_ATOMS * 100), 256, 0, stream>>>(agg2, (long)N_ATOMS * 100, ACT_RELU);
  bnstats_f32(agg2, 100, N_ATOMS, 100, g2g, g2b);
  k_bn<<<gs((long)N_ATOMS * 100), 256, 0, stream>>>(agg2, 100, scl, shf, agg2, nullptr, N_ATOMS, 100);

  // ---- segment max pooling -> drug_embed
  k_fill_u32<<<gs((long)NUM_DRUGS * 100), 256, 0, stream>>>(keys, 0u, (long)NUM_DRUGS * 100);
  k_segmax_enc<<<gs((long)N_ATOMS * 100), 256, 0, stream>>>(agg2, ibatch, keys, N_ATOMS, 100);
  k_segmax_dec<<<gs((long)NUM_DRUGS * 100), 256, 0, stream>>>(keys, demb, (long)NUM_DRUGS * 100);

  // ---- cell branch
  gemm(gexpr_bf, 672, wtc1, 672, nullptr, nullptr, bc1, C1, 128, nullptr, 0,
       N_CLINE, 128, 672, ACT_TANH);
  bnstats_f32(C1, 128, N_CLINE, 128, cg, cb);
  k_bn<<<gs((long)N_CLINE * 128), 256, 0, stream>>>(C1, 128, scl, shf, nullptr, C1bf, N_CLINE, 128);
  gemm(C1bf, 128, wtc2, 128, nullptr, nullptr, bc2, cellb, 100, nullptr, 0,
       N_CLINE, 100, 128, ACT_RELU);

  // ---- merge
  k_merge<<<gs((long)N_GRAPH * 128), 256, 0, stream>>>(demb, cellb, mergebf, 100, 128);

  // ---- hypergraph degrees
  const int* h_node = hedge;
  const int* h_he   = hedge + E_HYP;
  k_fill_f32<<<gs(N_GRAPH), 256, 0, stream>>>(Dd, 0.f, N_GRAPH);
  k_fill_f32<<<gs(N_HYPER), 256, 0, stream>>>(Bd, 0.f, N_HYPER);
  k_count<<<gs(E_HYP), 256, 0, stream>>>(h_node, Dd, E_HYP);
  k_count<<<gs(E_HYP), 256, 0, stream>>>(h_he, Bd, E_HYP);
  k_recip<<<gs(N_GRAPH), 256, 0, stream>>>(Dd, N_GRAPH);
  k_recip<<<gs(N_HYPER), 256, 0, stream>>>(Bd, N_HYPER);

  // ---- hyper conv 1
  gemm(mergebf, 128, wth1, 128, nullptr, nullptr, nullptr, HH, 256, nullptr, 0,
       N_GRAPH, 256, 128, ACT_NONE);
  k_fill_f32<<<gs((long)N_HYPER * 256), 256, 0, stream>>>(msg, 0.f, (long)N_HYPER * 256);
  k_hscat<<<gs((long)E_HYP * 256), 256, 0, stream>>>(h_node, h_he, HH, msg, E_HYP, 256);
  k_scalerows<<<gs((long)N_HYPER * 256), 256, 0, stream>>>(msg, Bd, N_HYPER, 256);
  k_fill_f32<<<gs((long)N_GRAPH * 256), 256, 0, stream>>>(aggH, 0.f, (long)N_GRAPH * 256);
  k_hscat<<<gs((long)E_HYP * 256), 256, 0, stream>>>(h_he, h_node, msg, aggH, E_HYP, 256);
  k_hfinish<<<gs((long)N_GRAPH * 256), 256, 0, stream>>>(aggH, Dd, bh1, N_GRAPH, 256);
  bnstats_f32(aggH, 256, N_GRAPH, 256, hg, hb);
  k_bn<<<gs((long)N_GRAPH * 256), 256, 0, stream>>>(aggH, 256, scl, shf, nullptr, hbf, N_GRAPH, 256);

  // ---- hyper conv 2 -> graph
  gemm(hbf, 256, wth2, 256, nullptr, nullptr, nullptr, HH, 256, nullptr, 0,
       N_GRAPH, 256, 256, ACT_NONE);
  k_fill_f32<<<gs((long)N_HYPER * 256), 256, 0, stream>>>(msg, 0.f, (long)N_HYPER * 256);
  k_hscat<<<gs((long)E_HYP * 256), 256, 0, stream>>>(h_node, h_he, HH, msg, E_HYP, 256);
  k_scalerows<<<gs((long)N_HYPER * 256), 256, 0, stream>>>(msg, Bd, N_HYPER, 256);
  k_fill_f32<<<gs((long)N_GRAPH * 256), 256, 0, stream>>>(aggG, 0.f, (long)N_GRAPH * 256);
  k_hscat<<<gs((long)E_HYP * 256), 256, 0, stream>>>(h_he, h_node, msg, aggG, E_HYP, 256);
  k_hfinish<<<gs((long)N_GRAPH * 256), 256, 0, stream>>>(aggG, Dd, bh2, N_GRAPH, 256);
  k_cvt_pad<<<gs((long)N_GRAPH * 256), 256, 0, stream>>>(aggG, gbf, N_GRAPH, 256, 256);

  // ---- reconstruction heads
  gemm(gbf, 256, wtrd, 256, nullptr, nullptr, nullptr, Td, 256, nullptr, 0,
       NUM_DRUGS, 256, 256, ACT_NONE);
  k_cvt_pad<<<gs((long)NUM_DRUGS * 256), 256, 0, stream>>>(Td, Tdbf, NUM_DRUGS, 256, 256);
  gemm(Tdbf, 256, gbf, 256, nullptr, nullptr, nullptr, out_rd, NUM_DRUGS, nullptr, 0,
       NUM_DRUGS, NUM_DRUGS, 256, ACT_SIGM);
  gemm(gbf + (size_t)NUM_DRUGS * 256, 256, wtrc, 256, nullptr, nullptr, nullptr,
       Tc, 256, nullptr, 0, N_CLINE, 256, 256, ACT_NONE);
  k_cvt_pad<<<gs((long)N_CLINE * 256), 256, 0, stream>>>(Tc, Tcbf, N_CLINE, 256, 256);
  gemm(Tcbf, 256, gbf + (size_t)NUM_DRUGS * 256, 256, nullptr, nullptr, nullptr,
       out_rc, N_CLINE, nullptr, 0, N_CLINE, N_CLINE, 256, ACT_SIGM);

  // ---- final MLP (gathered z -> z1 -> z2 -> res)
  {
    dim3 g(cdiv(384, 64), cdiv(N_IDX, 128));
    k_wmma_gather3<<<g, 256, 0, stream>>>(gbf, 256, indexp, wtd1, 768, bd1,
                                          z1, 384, N_IDX, 384, ACT_LEAKY);
  }
  bnstats_bf16(z1, 384, N_IDX, 384, d1g, d1b);
  gemm(z1, 384, wtd2, 384, scl, shf, bd2, nullptr, 0, z2, 192,
       N_IDX, 192, 384, ACT_LEAKY);
  bnstats_bf16(z2, 192, N_IDX, 192, d2g, d2b);
  k_final_dot<<<cdiv(N_IDX, 8), 256, 0, stream>>>(z2, scl, shf, Wd3, bd3, out_res, N_IDX);
}
